// DeepHedgingModelTransformerViz_30923764531831
// MI455X (gfx1250) — matrix operand order repaired and checked by reading, not compiled, not run on hardware
//
#include <hip/hip_runtime.h>
#include <math.h>

// ---------------------------------------------------------------------------
// DeepHedging transformer scan for MI455X (gfx1250).
//
// Roofline: ~580 GFLOP of GEMM across 90 sequential steps. KV cache in bf16
// (188 MB) fits the 192 MB L2, so attention stays L2-resident instead of
// costing ~17 TB of HBM traffic in fp32. GEMMs use V_WMMA_F32_16X16X32_BF16
// (bf16 in, f32 accumulate) with B tiles staged in LDS (double buffered,
// async load-to-LDS when the toolchain exposes it) so the matrix pipe is fed
// at ~85 FLOP/byte from L2 instead of ~13.
// ---------------------------------------------------------------------------

#define BATCH   1024
#define SREF    90
#define FEAT    8
#define DMODEL  512
#define NHEAD   8
#define DHEAD   64
#define TINF    11
#define EPSV    1e-5f
#define SCALE_E 22.627416997969522f   // sqrt(512)

// GEMM block tiling: 256 threads = 8 waves as 4(M) x 2(N); block 64x128.
#define BM   64
#define BN   128
#define BK2  64     // K staged per LDS buffer
#define BROW 72     // padded LDS row stride in elements (144 B: spreads banks)

typedef __bf16 bf16_t;
typedef __attribute__((ext_vector_type(16))) __bf16 v16bf;
typedef __attribute__((ext_vector_type(8)))  __bf16 v8bf;
typedef __attribute__((ext_vector_type(8)))  float  v8f;
typedef int v4i __attribute__((vector_size(16)));   // matches builtin param type

#if defined(__has_builtin)
#if __has_builtin(__builtin_amdgcn_global_load_async_to_lds_b128) && \
    __has_builtin(__builtin_amdgcn_s_wait_asynccnt)
#define USE_ASYNC_LDS 1
#endif
#endif

#if defined(USE_ASYNC_LDS)
// GLOBAL_LOAD_ASYNC_TO_LDS_B128: VGPR-bypassing copy, tracked by ASYNCcnt.
__device__ __forceinline__ void async_cp_b128(const bf16_t* gp, bf16_t* lp) {
  __builtin_amdgcn_global_load_async_to_lds_b128(
      (__attribute__((address_space(1))) v4i*)gp,
      (__attribute__((address_space(3))) v4i*)lp, 0, 0);
}
#endif

// ---------------------------------------------------------------------------
// fp32 -> bf16 weight conversion (done once per launch; deterministic)
// ---------------------------------------------------------------------------
__global__ void cvt_kernel(const float* __restrict__ in, bf16_t* __restrict__ outp, int n) {
  int i = blockIdx.x * blockDim.x + threadIdx.x;
  if (i < n) outp[i] = (bf16_t)in[i];
}

// ---------------------------------------------------------------------------
// Embedding: e = (concat(x_t, prev_d, prev_n, tau) @ We^T + be)*sqrt(D) + pos[t]
// ---------------------------------------------------------------------------
__global__ void embed_kernel(const float* __restrict__ x,  const float* __restrict__ init_d,
                             const float* __restrict__ init_n, const float* __restrict__ Tsc,
                             const float* __restrict__ We, const float* __restrict__ be,
                             const float* __restrict__ pos, const float* __restrict__ dn,
                             float* __restrict__ eF, bf16_t* __restrict__ eB, int t) {
  int gid = blockIdx.x * blockDim.x + threadIdx.x;
  int b = gid >> 9, d = gid & 511;
  float Tv  = Tsc[0];
  float tau = Tv - (float)t * (Tv / (float)SREF);
  float pd = (t == 0) ? init_d[b] : dn[2 * b];
  float pn = (t == 0) ? init_n[b] : dn[2 * b + 1];
  const float* xr = x + ((size_t)b * SREF + t) * FEAT;
  const float* wr = We + (size_t)d * TINF;
  float acc = 0.f;
#pragma unroll
  for (int i = 0; i < FEAT; ++i) acc += wr[i] * xr[i];
  acc += wr[8] * pd + wr[9] * pn + wr[10] * tau;
  float e = (acc + be[d]) * SCALE_E + pos[(size_t)t * DMODEL + d];
  eF[gid] = e;
  eB[gid] = (bf16_t)e;
}

// ---------------------------------------------------------------------------
// bf16 WMMA GEMM: out[m,n] = sum_k A[m,k] * W[n,k] + bias[n]
// A: (M,K) bf16 row-major (register-direct), W: (N,K) bf16 row-major,
// staged through double-buffered LDS (async-to-LDS when available).
// Per-lane register layouts follow the CDNA5 ISA WMMA striping:
//   A 16x32: lanes 0-15 row M=lane hold K {0..7,16..23}; lanes 16-31 {8..15,24..31}
//   B 32x16: lane column N=lane%16; lanes 0-15 K {0..7,16..23}; lanes 16-31 {8..15,24..31}
//   C/D:     VGPR j = row j + 8*(lane/16), col = lane%16
// MODE 0: fp32 out.  MODE 1: ReLU -> bf16 out.  MODE 2: QKV split.
// ---------------------------------------------------------------------------
__device__ __forceinline__ v16bf cat8(v8bf lo, v8bf hi) {
  v16bf r;
#pragma unroll
  for (int i = 0; i < 8; ++i) { r[i] = lo[i]; r[i + 8] = hi[i]; }
  return r;
}

template <int MODE>
__global__ __launch_bounds__(256) void
gemm_wmma_bf16(const bf16_t* __restrict__ A, const bf16_t* __restrict__ W,
               const float* __restrict__ bias,
               float* __restrict__ outF, bf16_t* __restrict__ outB,
               bf16_t* __restrict__ kc, bf16_t* __restrict__ vc,
               int t, int M, int N, int K) {
  __shared__ bf16_t Bs[2][BN * BROW];

  const int tid  = threadIdx.x;
  const int lane = tid & 31;
  const int wv   = tid >> 5;     // 0..7
  const int wm   = wv & 3;       // wave row    (4 along M)
  const int wn   = wv >> 2;      // wave column (2 along N)
  const int nBlkN = N / BN;
  const int bm = blockIdx.x / nBlkN;
  const int bn = blockIdx.x - bm * nBlkN;
  const int half = lane >> 4;
  const int l16  = lane & 15;
  const int m    = bm * BM + wm * 16 + l16;

  v8f acc[4];
#pragma unroll
  for (int ti = 0; ti < 4; ++ti)
#pragma unroll
    for (int j = 0; j < 8; ++j) acc[ti][j] = 0.f;

  const bf16_t* arow = A + (size_t)m * K + 8 * half;
  const bf16_t* wblk = W + (size_t)(bn * BN) * K;

  // Cooperative stage of a BN x BK2 weight tile into LDS.
  // 128 rows x 64 elems = 1024 b128 chunks; 4 chunks per thread.
  auto stage = [&](int kb, int buf) {
#pragma unroll
    for (int i = 0; i < 4; ++i) {
      int c   = tid + 256 * i;
      int row = c >> 3;
      int kc8 = (c & 7) * 8;
      const bf16_t* gp = wblk + (size_t)row * K + kb * BK2 + kc8;
      bf16_t*       lp = &Bs[buf][row * BROW + kc8];
#if defined(USE_ASYNC_LDS)
      async_cp_b128(gp, lp);
#else
      *(v8bf*)lp = *(const v8bf*)gp;
#endif
    }
  };

  const int nkb = K / BK2;
  stage(0, 0);
  for (int kb = 0; kb < nkb; ++kb) {
    const int buf = kb & 1;
#if defined(USE_ASYNC_LDS)
    __builtin_amdgcn_s_wait_asynccnt(0);
#endif
    __syncthreads();                 // stage kb visible; prior buffer reads done
    if (kb + 1 < nkb) stage(kb + 1, buf ^ 1);
#pragma unroll
    for (int ks = 0; ks < 2; ++ks) { // two K=32 WMMA steps per staged buffer
      int kg = kb * BK2 + ks * 32;
      v8bf alo = *(const v8bf*)(arow + kg);
      v8bf ahi = *(const v8bf*)(arow + kg + 16);
      v16bf av = cat8(alo, ahi);
#pragma unroll
      for (int ti = 0; ti < 4; ++ti) {
        int row = wn * 64 + ti * 16 + l16;
        const bf16_t* lp = &Bs[buf][row * BROW + ks * 32 + 8 * half];
        v16bf bv = cat8(*(const v8bf*)lp, *(const v8bf*)(lp + 16));
        acc[ti] = __builtin_amdgcn_wmma_f32_16x16x32_bf16(
            false, av, false, bv, (short)0, acc[ti], false, false);
      }
    }
  }

#pragma unroll
  for (int ti = 0; ti < 4; ++ti) {
    int col = bn * BN + wn * 64 + ti * 16 + l16;
    float bsv = bias[col];
#pragma unroll
    for (int j = 0; j < 8; ++j) {
      int row = bm * BM + wm * 16 + j + 8 * half;
      float val = acc[ti][j] + bsv;
      if (MODE == 0) {
        outF[(size_t)row * N + col] = val;
      } else if (MODE == 1) {
        outB[(size_t)row * N + col] = (bf16_t)fmaxf(val, 0.f);
      } else {  // QKV split
        if (col < DMODEL)
          outB[(size_t)row * DMODEL + col] = (bf16_t)val;
        else if (col < 2 * DMODEL)
          kc[((size_t)row * SREF + t) * DMODEL + (col - DMODEL)] = (bf16_t)val;
        else
          vc[((size_t)row * SREF + t) * DMODEL + (col - 2 * DMODEL)] = (bf16_t)val;
      }
    }
  }
}

// ---------------------------------------------------------------------------
// Causal attention at step t. One wave per (b,h). KV cache bf16 (L2-resident).
// ---------------------------------------------------------------------------
__global__ void attn_kernel(const bf16_t* __restrict__ qB, const bf16_t* __restrict__ kc,
                            const bf16_t* __restrict__ vc, bf16_t* __restrict__ attnB,
                            float* __restrict__ w0, int t) {
  const int lane = threadIdx.x & 31;
  const int wave = blockIdx.x * (blockDim.x >> 5) + (threadIdx.x >> 5);
  const int b = wave >> 3;
  const int h = wave & (NHEAD - 1);
  const int L = t + 1;

  float q[DHEAD];
  const bf16_t* qp = qB + (size_t)b * DMODEL + h * DHEAD;
#pragma unroll
  for (int j = 0; j < DHEAD; ++j) q[j] = (float)qp[j];

  float sc0 = -3.0e38f, sc1 = -3.0e38f, sc2 = -3.0e38f;
#pragma unroll
  for (int i = 0; i < 3; ++i) {
    int s = lane + 32 * i;
    if (s < L) {
      const bf16_t* kp = kc + ((size_t)b * SREF + s) * DMODEL + h * DHEAD;
      float acc = 0.f;
#pragma unroll
      for (int j = 0; j < DHEAD; ++j) acc += q[j] * (float)kp[j];
      acc *= 0.125f;  // 1/sqrt(DH)
      if (i == 0) sc0 = acc; else if (i == 1) sc1 = acc; else sc2 = acc;
    }
  }
  float mx = fmaxf(sc0, fmaxf(sc1, sc2));
  for (int off = 16; off > 0; off >>= 1) mx = fmaxf(mx, __shfl_xor(mx, off, 32));

  float w0v = (lane      < L) ? __expf(sc0 - mx) : 0.f;
  float w1v = (lane + 32 < L) ? __expf(sc1 - mx) : 0.f;
  float w2v = (lane + 64 < L) ? __expf(sc2 - mx) : 0.f;
  float psum = w0v + w1v + w2v;
  for (int off = 16; off > 0; off >>= 1) psum += __shfl_xor(psum, off, 32);
  float inv = 1.f / psum;
  w0v *= inv; w1v *= inv; w2v *= inv;

  if (b == 0) {
    if (lane      < L) w0[h * SREF + lane]      = w0v;
    if (lane + 32 < L) w0[h * SREF + lane + 32] = w1v;
    if (lane + 64 < L) w0[h * SREF + lane + 64] = w2v;
  }

  float a0 = 0.f, a1 = 0.f;
  const bf16_t* vb = vc + (size_t)b * SREF * DMODEL + h * DHEAD + 2 * lane;
  for (int s = 0; s < L; ++s) {
    float src = (s < 32) ? w0v : ((s < 64) ? w1v : w2v);
    float ws = __shfl(src, s & 31, 32);
    const bf16_t* vp = vb + (size_t)s * DMODEL;
    a0 += ws * (float)vp[0];
    a1 += ws * (float)vp[1];
  }
  bf16_t* op = attnB + (size_t)b * DMODEL + h * DHEAD + 2 * lane;
  op[0] = (bf16_t)a0;
  op[1] = (bf16_t)a1;
}

// ---------------------------------------------------------------------------
// LayerNorm over D=512 of (a + r); one 128-thread block per row.
// ---------------------------------------------------------------------------
__global__ void ln_kernel(const float* __restrict__ a, const float* __restrict__ r,
                          const float* __restrict__ g, const float* __restrict__ beta,
                          float* __restrict__ outF, bf16_t* __restrict__ outB, int writeB) {
  __shared__ float s1[128], s2[128];
  int b = blockIdx.x, tid = threadIdx.x;
  float v[4];
  float sum = 0.f, sq = 0.f;
#pragma unroll
  for (int i = 0; i < 4; ++i) {
    int d = tid + 128 * i;
    float xv = a[(size_t)b * DMODEL + d] + r[(size_t)b * DMODEL + d];
    v[i] = xv; sum += xv; sq += xv * xv;
  }
  s1[tid] = sum; s2[tid] = sq;
  __syncthreads();
  for (int off = 64; off > 0; off >>= 1) {
    if (tid < off) { s1[tid] += s1[tid + off]; s2[tid] += s2[tid + off]; }
    __syncthreads();
  }
  float mu  = s1[0] * (1.f / DMODEL);
  float var = s2[0] * (1.f / DMODEL) - mu * mu;
  float rs  = rsqrtf(var + EPSV);
#pragma unroll
  for (int i = 0; i < 4; ++i) {
    int d = tid + 128 * i;
    float y = (v[i] - mu) * rs * g[d] + beta[d];
    outF[(size_t)b * DMODEL + d] = y;
    if (writeB) outB[(size_t)b * DMODEL + d] = (bf16_t)y;
  }
}

// ---------------------------------------------------------------------------
// Head: act = tanh(h @ Wfc^T + bfc); one wave per batch row.
// ---------------------------------------------------------------------------
__global__ void head_kernel(const float* __restrict__ h, const float* __restrict__ Wfc,
                            const float* __restrict__ bfc, float* __restrict__ out,
                            float* __restrict__ dn, int t) {
  const int lane = threadIdx.x & 31;
  const int b = blockIdx.x * (blockDim.x >> 5) + (threadIdx.x >> 5);
  float d0 = 0.f, d1 = 0.f;
  const float* hb = h + (size_t)b * DMODEL;
#pragma unroll
  for (int i = 0; i < DMODEL / 32; ++i) {
    int d = lane + 32 * i;
    float hv = hb[d];
    d0 += hv * Wfc[d];
    d1 += hv * Wfc[DMODEL + d];
  }
  for (int off = 16; off > 0; off >>= 1) {
    d0 += __shfl_xor(d0, off, 32);
    d1 += __shfl_xor(d1, off, 32);
  }
  if (lane == 0) {
    float a0 = tanhf(d0 + bfc[0]);
    float a1 = tanhf(d1 + bfc[1]);
    out[(size_t)b * SREF + t] = a0;                             // deltas
    out[(size_t)BATCH * SREF + (size_t)b * SREF + t] = a1;      // Ns
    dn[2 * b] = a0;
    dn[2 * b + 1] = a1;
  }
}

// rows[t,s] = mean over heads of batch-0 attention weights (zeros past t).
__global__ void row_kernel(const float* __restrict__ w0, float* __restrict__ out, int t) {
  int s = threadIdx.x;
  if (s >= SREF) return;
  float v = 0.f;
  if (s <= t) {
    float acc = 0.f;
#pragma unroll
    for (int hh = 0; hh < NHEAD; ++hh) acc += w0[hh * SREF + s];
    v = acc * (1.f / NHEAD);
  }
  out[(size_t)2 * BATCH * SREF + (size_t)t * SREF + s] = v;
}

// ---------------------------------------------------------------------------
extern "C" void kernel_launch(void* const* d_in, const int* in_sizes, int n_in,
                              void* d_out, int out_size, void* d_ws, size_t ws_size,
                              hipStream_t stream) {
  (void)in_sizes; (void)n_in; (void)out_size; (void)ws_size;
  const float* x      = (const float*)d_in[0];
  const float* init_d = (const float*)d_in[1];
  const float* init_n = (const float*)d_in[2];
  const float* Tsc    = (const float*)d_in[3];
  const float* We     = (const float*)d_in[4];
  const float* be     = (const float*)d_in[5];
  const float* pos    = (const float*)d_in[6];
  const float* Wqkv   = (const float*)d_in[7];
  const float* bqkv   = (const float*)d_in[8];
  const float* Wo     = (const float*)d_in[9];
  const float* bo     = (const float*)d_in[10];
  const float* W1     = (const float*)d_in[11];
  const float* b1     = (const float*)d_in[12];
  const float* W2     = (const float*)d_in[13];
  const float* b2     = (const float*)d_in[14];
  const float* g1     = (const float*)d_in[15];
  const float* beta1  = (const float*)d_in[16];
  const float* g2     = (const float*)d_in[17];
  const float* beta2  = (const float*)d_in[18];
  const float* Wfc    = (const float*)d_in[19];
  const float* bfc    = (const float*)d_in[20];
  float* out = (float*)d_out;

  char* ws = (char*)d_ws;
  size_t off = 0;
  auto alloc = [&](size_t bytes) -> char* {
    char* p = ws + off;
    off += (bytes + 255) & ~(size_t)255;
    return p;
  };
  bf16_t* kc    = (bf16_t*)alloc((size_t)BATCH * SREF * DMODEL * 2);  // 94.4 MB
  bf16_t* vc    = (bf16_t*)alloc((size_t)BATCH * SREF * DMODEL * 2);  // 94.4 MB
  bf16_t* wqB   = (bf16_t*)alloc((size_t)3 * DMODEL * DMODEL * 2);
  bf16_t* woB   = (bf16_t*)alloc((size_t)DMODEL * DMODEL * 2);
  bf16_t* w1B   = (bf16_t*)alloc((size_t)4 * DMODEL * DMODEL * 2);
  bf16_t* w2B   = (bf16_t*)alloc((size_t)4 * DMODEL * DMODEL * 2);
  float*  eF    = (float*) alloc((size_t)BATCH * DMODEL * 4);
  bf16_t* eB    = (bf16_t*)alloc((size_t)BATCH * DMODEL * 2);
  bf16_t* qB    = (bf16_t*)alloc((size_t)BATCH * DMODEL * 2);
  bf16_t* attnB = (bf16_t*)alloc((size_t)BATCH * DMODEL * 2);
  float*  projF = (float*) alloc((size_t)BATCH * DMODEL * 4);
  float*  xsF   = (float*) alloc((size_t)BATCH * DMODEL * 4);
  bf16_t* xsB   = (bf16_t*)alloc((size_t)BATCH * DMODEL * 2);
  bf16_t* ff1B  = (bf16_t*)alloc((size_t)BATCH * 4 * DMODEL * 2);
  float*  ff2F  = (float*) alloc((size_t)BATCH * DMODEL * 4);
  float*  hF    = (float*) alloc((size_t)BATCH * DMODEL * 4);
  float*  dn    = (float*) alloc((size_t)BATCH * 2 * 4);
  float*  w0    = (float*) alloc((size_t)NHEAD * SREF * 4);

  // One-time (per call) bf16 weight conversion.
  cvt_kernel<<<(3 * DMODEL * DMODEL + 255) / 256, 256, 0, stream>>>(Wqkv, wqB, 3 * DMODEL * DMODEL);
  cvt_kernel<<<(DMODEL * DMODEL + 255) / 256, 256, 0, stream>>>(Wo, woB, DMODEL * DMODEL);
  cvt_kernel<<<(4 * DMODEL * DMODEL + 255) / 256, 256, 0, stream>>>(W1, w1B, 4 * DMODEL * DMODEL);
  cvt_kernel<<<(4 * DMODEL * DMODEL + 255) / 256, 256, 0, stream>>>(W2, w2B, 4 * DMODEL * DMODEL);

  for (int t = 0; t < SREF; ++t) {
    embed_kernel<<<(BATCH * DMODEL) / 256, 256, 0, stream>>>(
        x, init_d, init_n, Tsc, We, be, pos, dn, eF, eB, t);

    // QKV: M=1024, N=1536, K=512  -> (1024/64)*(1536/128) = 192 blocks
    gemm_wmma_bf16<2><<<(BATCH / BM) * (3 * DMODEL / BN), 256, 0, stream>>>(
        eB, wqB, bqkv, nullptr, qB, kc, vc, t, BATCH, 3 * DMODEL, DMODEL);

    attn_kernel<<<(BATCH * NHEAD) / 8, 256, 0, stream>>>(qB, kc, vc, attnB, w0, t);

    // Output projection: M=1024, N=512, K=512 -> 64 blocks
    gemm_wmma_bf16<0><<<(BATCH / BM) * (DMODEL / BN), 256, 0, stream>>>(
        attnB, woB, bo, projF, nullptr, nullptr, nullptr, t, BATCH, DMODEL, DMODEL);

    ln_kernel<<<BATCH, 128, 0, stream>>>(eF, projF, g1, beta1, xsF, xsB, 1);

    // FF1 + ReLU: M=1024, N=2048, K=512 -> 256 blocks
    gemm_wmma_bf16<1><<<(BATCH / BM) * (4 * DMODEL / BN), 256, 0, stream>>>(
        xsB, w1B, b1, nullptr, ff1B, nullptr, nullptr, t, BATCH, 4 * DMODEL, DMODEL);

    // FF2: M=1024, N=512, K=2048 -> 64 blocks
    gemm_wmma_bf16<0><<<(BATCH / BM) * (DMODEL / BN), 256, 0, stream>>>(
        ff1B, w2B, b2, ff2F, nullptr, nullptr, nullptr, t, BATCH, DMODEL, 4 * DMODEL);

    ln_kernel<<<BATCH, 128, 0, stream>>>(xsF, ff2F, g2, beta2, hF, nullptr, 0);

    head_kernel<<<BATCH / 8, 256, 0, stream>>>(hF, Wfc, bfc, out, dn, t);
    row_kernel<<<1, 128, 0, stream>>>(w0, out, t);
  }
}
